// GKTGNN_63737314673104
// MI455X (gfx1250) — matrix operand order repaired and checked by reading, not compiled, not run on hardware
//
#include <hip/hip_runtime.h>
#include <hip/hip_bf16.h>

#define N_NODES   50000
#define N_EDGES   800000
#define EMBED_DIM 256
#define HIDDEN    128
#define OUT_DIM   12
#define N_TARGETS 4096

typedef __attribute__((ext_vector_type(16))) __bf16          v16bf;
typedef __attribute__((ext_vector_type(8)))  float           v8f;
typedef __attribute__((ext_vector_type(4)))  unsigned int    v4u;
typedef __attribute__((ext_vector_type(8)))  unsigned int    v8u;

__device__ __forceinline__ unsigned short f2bf(float f) {
    unsigned int u = __builtin_bit_cast(unsigned int, f);
    unsigned int r = u + 0x7FFFu + ((u >> 16) & 1u);   // round-to-nearest-even
    return (unsigned short)(r >> 16);
}

__device__ __forceinline__ v16bf load_frag(const unsigned short* p0,
                                           const unsigned short* p1) {
    v4u lo = *(const v4u*)p0;           // 8 bf16 (global_load_b128)
    v4u hi = *(const v4u*)p1;           // 8 bf16
    v8u m  = __builtin_shufflevector(lo, hi, 0, 1, 2, 3, 4, 5, 6, 7);
    return __builtin_bit_cast(v16bf, m);
}

// ---------------------------------------------------------------------------
// float -> bf16 row-major copy (feeds WMMA A operand directly)
// ---------------------------------------------------------------------------
__global__ void cvt_f32_bf16(const float* __restrict__ in,
                             unsigned short* __restrict__ out, int n) {
    int i = blockIdx.x * blockDim.x + threadIdx.x;
    if (i < n) out[i] = f2bf(in[i]);
}

// ---------------------------------------------------------------------------
// Pack W[K,128] fp32 into bf16 fragment order:
// idx = ((kc*8 + nt)*32 + lane)*16 + j ;  k = kc*32 + (lane>>4)*16 + j ;
// n = nt*16 + (lane&15)
// ---------------------------------------------------------------------------
__global__ void pack_w_bf16(const float* __restrict__ W,
                            unsigned short* __restrict__ Wp, int K) {
    int idx = blockIdx.x * blockDim.x + threadIdx.x;
    if (idx >= (K / 32) * 4096) return;
    int j  = idx & 15;
    int l  = (idx >> 4) & 31;
    int nt = (idx >> 9) & 7;
    int kc = idx >> 12;
    int k  = kc * 32 + (l >> 4) * 16 + j;
    int n  = nt * 16 + (l & 15);
    Wp[idx] = f2bf(W[k * HIDDEN + n]);
}

// ---------------------------------------------------------------------------
// WMMA GEMM: C[M,128] = A[M,K](bf16 row-major) @ Wp(prepacked bf16)
// Block = 256 threads = 8 waves; block -> 16-row M tile; wave w -> N tile w.
// K compile-time -> fully unrolled WMMA chain. M % 16 == 0.
// ---------------------------------------------------------------------------
template <int K>
__global__ void __launch_bounds__(256)
gcn_gemm_wmma(const unsigned short* __restrict__ Abf,
              const unsigned short* __restrict__ Wp,
              float* __restrict__ C) {
    const int wave = threadIdx.x >> 5;          // 0..7 -> N tile
    const int lane = threadIdx.x & 31;
    const int row  = lane & 15;
    const int half = lane >> 4;
    const size_t m0 = (size_t)blockIdx.x * 16;

    const unsigned short* arow = Abf + (m0 + row) * (size_t)K;
    const unsigned short* bptr = Wp + (size_t)wave * 512 + (size_t)lane * 16;

    v8f acc = {};
    #pragma unroll
    for (int kc = 0; kc < K / 32; ++kc) {
        // A: elems 0..15 -> k in [kc*32+half*16, +16)
        const unsigned short* ak = arow + kc * 32 + half * 16;
        v16bf av = load_frag(ak, ak + 8);
        // B: prepacked, 32 contiguous bytes per lane per chunk
        const unsigned short* bk = bptr + (size_t)kc * 4096;
        v16bf bv = load_frag(bk, bk + 8);
        acc = __builtin_amdgcn_wmma_f32_16x16x32_bf16(
            false, av, false, bv, (short)0, acc, false, false);
    }

    // C/D layout: VGPR r -> M = half*8 + r, N = lane&15
    #pragma unroll
    for (int r = 0; r < 8; ++r)
        C[(m0 + half * 8 + r) * (size_t)HIDDEN + wave * 16 + row] = acc[r];
}

// ---------------------------------------------------------------------------
__global__ void fill_f32(float* __restrict__ p, float v, int n) {
    int i = blockIdx.x * blockDim.x + threadIdx.x;
    if (i < n) p[i] = v;
}

__global__ void deg_count(const int* __restrict__ ei, float* __restrict__ deg) {
    int e = blockIdx.x * blockDim.x + threadIdx.x;
    if (e < N_EDGES) atomicAdd(&deg[ei[N_EDGES + e]], 1.0f);
}

__global__ void deg_to_dinv(float* __restrict__ d) {
    int i = blockIdx.x * blockDim.x + threadIdx.x;
    if (i < N_NODES) d[i] = __frsqrt_rn(d[i]);      // deg >= 1 (self loop)
}

// 32 lanes per edge, 4 floats per lane (HIDDEN = 128)
__global__ void __launch_bounds__(256)
edge_agg(const float* __restrict__ h, const int* __restrict__ ei,
         const float* __restrict__ dinv, float* __restrict__ agg) {
    int t    = blockIdx.x * blockDim.x + threadIdx.x;
    int edge = t >> 5;
    int part = t & 31;
    if (edge >= N_EDGES) return;
    int s = ei[edge];
    int d = ei[N_EDGES + edge];
    float nrm = dinv[s] * dinv[d];
    float4 v = *(const float4*)(h + (size_t)s * HIDDEN + part * 4);
    float* out = agg + (size_t)d * HIDDEN + part * 4;
    atomicAdd(out + 0, v.x * nrm);
    atomicAdd(out + 1, v.y * nrm);
    atomicAdd(out + 2, v.z * nrm);
    atomicAdd(out + 3, v.w * nrm);
}

// out[i] = relu(agg[i] + h[i]*dinv[node]^2 + bias[c])   (self-loop fused)
__global__ void epilogue(const float* __restrict__ agg, const float* __restrict__ h,
                         const float* __restrict__ dinv, const float* __restrict__ bias,
                         float* __restrict__ out) {
    int i = blockIdx.x * blockDim.x + threadIdx.x;
    if (i >= N_NODES * HIDDEN) return;
    int node = i >> 7;
    int c    = i & (HIDDEN - 1);
    float dv = dinv[node];
    float v  = agg[i] + h[i] * dv * dv + bias[c];
    out[i] = fmaxf(v, 0.0f);
}

// pred[t,o] = h[mask[t]] . Wr[:,o] + br[o]
__global__ void readout(const float* __restrict__ h, const int* __restrict__ mask,
                        const float* __restrict__ Wr, const float* __restrict__ br,
                        float* __restrict__ out) {
    int t = blockIdx.x;
    int o = threadIdx.x;
    if (o >= OUT_DIM) return;
    const float* hr = h + (size_t)mask[t] * HIDDEN;
    float acc = br[o];
    #pragma unroll 4
    for (int c = 0; c < HIDDEN; ++c) acc += hr[c] * Wr[c * OUT_DIM + o];
    out[t * OUT_DIM + o] = acc;
}

// ---------------------------------------------------------------------------
extern "C" void kernel_launch(void* const* d_in, const int* in_sizes, int n_in,
                              void* d_out, int out_size, void* d_ws, size_t ws_size,
                              hipStream_t stream) {
    const float* x    = (const float*)d_in[0];
    const int*   ei   = (const int*)  d_in[1];
    const int*   mask = (const int*)  d_in[2];
    const float* W1   = (const float*)d_in[3];
    const float* b1   = (const float*)d_in[4];
    const float* W2   = (const float*)d_in[5];
    const float* b2   = (const float*)d_in[6];
    const float* Wr   = (const float*)d_in[7];
    const float* br   = (const float*)d_in[8];
    float* out = (float*)d_out;

    // workspace carve-up (256B aligned)
    char* ws = (char*)d_ws;
    size_t off = 0;
    auto carve = [&](size_t bytes) -> char* {
        char* p = ws + off;
        off = (off + bytes + 255) & ~(size_t)255;
        return p;
    };
    const size_t NODE_F = (size_t)N_NODES * HIDDEN;                  // 6.4M
    float*          dinv = (float*)         carve(N_NODES * 4);
    unsigned short* xbf  = (unsigned short*)carve((size_t)N_NODES * EMBED_DIM * 2);
    unsigned short* hbf  = (unsigned short*)carve(NODE_F * 2);
    unsigned short* W1p  = (unsigned short*)carve((size_t)EMBED_DIM * HIDDEN * 2);
    unsigned short* W2p  = (unsigned short*)carve((size_t)HIDDEN * HIDDEN * 2);
    float*          bufA = (float*)         carve(NODE_F * 4);
    float*          bufB = (float*)         carve(NODE_F * 4);
    float*          bufC = (float*)         carve(NODE_F * 4);

    const int B = 256;
    dim3 blk(B);
    auto nb = [](size_t n) { return (unsigned)((n + 255) / 256); };

    // 1) degree (incl. self loop) -> dinv, in place
    fill_f32<<<nb(N_NODES), blk, 0, stream>>>(dinv, 1.0f, N_NODES);
    deg_count<<<nb(N_EDGES), blk, 0, stream>>>(ei, dinv);
    deg_to_dinv<<<nb(N_NODES), blk, 0, stream>>>(dinv);

    // 2) precision prep: x -> bf16, pack W1/W2
    cvt_f32_bf16<<<nb((size_t)N_NODES * EMBED_DIM), blk, 0, stream>>>(
        x, xbf, N_NODES * EMBED_DIM);
    pack_w_bf16<<<nb(EMBED_DIM * HIDDEN), blk, 0, stream>>>(W1, W1p, EMBED_DIM);
    pack_w_bf16<<<nb(HIDDEN * HIDDEN), blk, 0, stream>>>(W2, W2p, HIDDEN);

    // 3) layer 1: h1 = x @ W1  (K=256)
    gcn_gemm_wmma<EMBED_DIM><<<N_NODES / 16, blk, 0, stream>>>(xbf, W1p, bufA);
    fill_f32<<<nb(NODE_F), blk, 0, stream>>>(bufB, 0.0f, (int)NODE_F);
    edge_agg<<<nb((size_t)N_EDGES * 32), blk, 0, stream>>>(bufA, ei, dinv, bufB);
    epilogue<<<nb(NODE_F), blk, 0, stream>>>(bufB, bufA, dinv, b1, bufC);

    // 4) layer 2: h2 = h1r @ W2  (K=128)
    cvt_f32_bf16<<<nb(NODE_F), blk, 0, stream>>>(bufC, hbf, (int)NODE_F);
    gcn_gemm_wmma<HIDDEN><<<N_NODES / 16, blk, 0, stream>>>(hbf, W2p, bufA);
    fill_f32<<<nb(NODE_F), blk, 0, stream>>>(bufB, 0.0f, (int)NODE_F);
    edge_agg<<<nb((size_t)N_EDGES * 32), blk, 0, stream>>>(bufA, ei, dinv, bufB);
    epilogue<<<nb(NODE_F), blk, 0, stream>>>(bufB, bufA, dinv, b2, bufC);

    // 5) readout
    readout<<<N_TARGETS, dim3(32), 0, stream>>>(bufC, mask, Wr, br, out);
}
